// Model_12206297055798
// MI455X (gfx1250) — compile-verified, matrix operand order checked by reading, not run on hardware
//
#include <hip/hip_runtime.h>
#include <math.h>

#define NROWS 150000
#define EPSV  1e-5f

typedef __attribute__((ext_vector_type(16))) _Float16     v16h;
typedef __attribute__((ext_vector_type(8)))  float        v8f;
typedef __attribute__((ext_vector_type(4)))  unsigned int u32x4;
typedef __attribute__((ext_vector_type(8)))  int          i32x8;
typedef __attribute__((ext_vector_type(4)))  int          i32x4;

#if defined(__AMDGCN__) && __has_builtin(__builtin_amdgcn_tensor_load_to_lds)
#define USE_TDM 1
#else
#define USE_TDM 0
#endif

#if USE_TDM
// Issue one 2-D TDM tile load: 16 rows x 32 cols of f32, row stride = lda elems.
// D# layout per CDNA5 ISA sec 8.3-8.6. Tensor dims set huge (tiles always in
// bounds); global_addr is the tile start; groups 2/3 zero (tile_dim2=0 unused).
// This toolchain exposes the 6-arg builtin: (g0 u32x4, g1 i32x8, g2 i32x4,
// g3 i32x4, extra i32x8, cpol i32).
__device__ __forceinline__ void tdm_load_tile(unsigned lds_off, const float* g, int lda)
{
    const unsigned long long ga = (unsigned long long)(uintptr_t)g;
    u32x4 g0;
    g0[0] = 1u;                                        // count=1 (valid user D#)
    g0[1] = lds_off;                                   // LDS byte address
    g0[2] = (unsigned)ga;                              // global_addr[31:0]
    g0[3] = (unsigned)((ga >> 32) & 0x01FFFFFFu) | (2u << 30);   // [56:32] | type=2
    i32x8 g1;
    g1[0] = (int)(2u << 16);                           // wg_mask=0, data_size=2 (4B)
    g1[1] = 0;                                         // tensor_dim0 = 1<<30 (lo16=0)
    g1[2] = (int)0x4000u;                              // dim0 hi16 ; tensor_dim1 lo16=0
    g1[3] = (int)(0x4000u | (32u << 16));              // dim1 hi16 ; tile_dim0 = 32
    g1[4] = 16;                                        // tile_dim1 = 16, tile_dim2 = 0
    g1[5] = lda;                                       // tensor_dim0_stride (elems)
    g1[6] = 0;
    g1[7] = 0;
    i32x4 z4; z4[0] = 0; z4[1] = 0; z4[2] = 0; z4[3] = 0;
    i32x8 z8;
    #pragma unroll
    for (int i = 0; i < 8; ++i) z8[i] = 0;
    __builtin_amdgcn_tensor_load_to_lds(g0, g1, z4, z4, z8, 0);
}
#endif

// ---------------------------------------------------------------------------
// WMMA GEMM:  C[M x N] = act( concat(A0|A1|A2)[M x Ktot] @ W[Ktot x N] + bias )
//   - 128 threads = 4 wave32; each wave owns one 16-row tile and all N cols
//   - W converted f32->f16 into LDS in *B-fragment order* so each lane's 16
//     fragment halves are one contiguous 32B run (2x ds_load_b128, no strided
//     ds_load_u16 scalarization)
//   - A tiles staged by the Tensor Data Mover (double buffered, TENSORcnt
//     pipelined); manual-staging fallback if the builtin is unavailable
//   - act: 0 = none, 1 = tanh
// ---------------------------------------------------------------------------
template<int NT>   // NT = N/16  (4 or 8)
__global__ __launch_bounds__(128) void gemm_wmma(
    const float* __restrict__ A0, int K0, int lda0,
    const float* __restrict__ A1, int K1, int lda1,
    const float* __restrict__ A2, int K2, int lda2,
    const float* __restrict__ W,  const float* __restrict__ bias,
    float* __restrict__ C, int ldc, int M, int act)
{
    const int N    = NT * 16;
    const int Ktot = K0 + K1 + K2;

    __shared__ __align__(32) _Float16 sW[16384];      // weights, fragment order (32 KB)
    __shared__ __align__(16) float    sA[4][2][16*32];// per-wave dbl-buffered tiles (16 KB)

    const int tid = threadIdx.x;
    // Cooperative swizzled fill: sW[((kt*NT+t)*32 + lane)*16 + j] =
    //   (f16) W[(kt*32 + kf(j,lane>>4)) * N + t*16 + (lane&15)]
    for (int i = tid; i < Ktot * N; i += 128) {
        const int j     = i & 15;
        const int lane2 = (i >> 4) & 31;
        const int rest  = i >> 9;            // kt*NT + t
        const int t     = rest % NT;
        const int kt    = rest / NT;
        const int kf    = j + 8 * (lane2 >> 4) + ((j >= 8) ? 8 : 0);
        sW[i] = (_Float16)W[(kt * 32 + kf) * N + t * 16 + (lane2 & 15)];
    }
    __syncthreads();

    const int wave    = tid >> 5;
    const int lane    = tid & 31;
    const int tileM   = blockIdx.x * 4 + wave;
    const int rowbase = tileM * 16;
    if (rowbase >= M) return;                // M % 16 == 0

    v8f acc[NT];
    #pragma unroll
    for (int t = 0; t < NT; ++t)
        #pragma unroll
        for (int e = 0; e < 8; ++e) acc[t][e] = 0.0f;

    const int halfsel = lane >> 4;
    const int nlane   = lane & 15;
    const int nk      = Ktot >> 5;

    auto seg_ptr = [&](int k, int& lda) -> const float* {
        if (k < K0)      { lda = lda0; return A0 + (size_t)rowbase * lda0 + k; }
        if (k < K0 + K1) { lda = lda1; return A1 + (size_t)rowbase * lda1 + (k - K0); }
        lda = lda2; return A2 + (size_t)rowbase * lda2 + (k - K0 - K1);
    };

#if USE_TDM
    {   // prologue: kick off tile 0
        int lda; const float* g = seg_ptr(0, lda);
        tdm_load_tile((unsigned)(uintptr_t)&sA[wave][0][0], g, lda);
    }
#endif

    for (int kt = 0; kt < nk; ++kt) {
        const int buf = kt & 1;
#if USE_TDM
        if (kt + 1 < nk) {   // issue next tile, then wait for current one
            int lda; const float* g = seg_ptr((kt + 1) << 5, lda);
            tdm_load_tile((unsigned)(uintptr_t)&sA[wave][buf ^ 1][0], g, lda);
            __builtin_amdgcn_s_wait_tensorcnt(1);
        } else {
            __builtin_amdgcn_s_wait_tensorcnt(0);
        }
        __builtin_amdgcn_wave_barrier();
#else
        {   // manual staging: each lane copies half a row (16 f32) of the tile
            int lda; const float* gp0 = seg_ptr(kt << 5, lda);
            const int r  = lane >> 1;
            const int c0 = (lane & 1) * 16;
            const float* gp = gp0 + (size_t)r * lda + c0;
            float* s = &sA[wave][buf][r * 32 + c0];
            #pragma unroll
            for (int v = 0; v < 4; ++v)
                *(float4*)(s + v * 4) = *(const float4*)(gp + v * 4);
        }
        __builtin_amdgcn_wave_barrier();     // same-wave DS ops stay in order
#endif
        // A fragment: two contiguous 8-elem runs per lane, f32 -> f16
        v16h a;
        {
            const float* s = &sA[wave][buf][0];
            #pragma unroll
            for (int j = 0; j < 16; ++j) {
                const int kf = j + 8 * halfsel + ((j >= 8) ? 8 : 0);
                a[j] = (_Float16)s[nlane * 32 + kf];
            }
        }
        // B fragments: one contiguous 32B run per lane (pre-swizzled)
        #pragma unroll
        for (int t = 0; t < NT; ++t) {
            const v16h b = *(const v16h*)&sW[((kt * NT + t) * 32 + lane) * 16];
            acc[t] = __builtin_amdgcn_wmma_f32_16x16x32_f16(
                false, a, false, b, (short)0, acc[t], false, false);
        }
        __builtin_amdgcn_wave_barrier();
    }

    // epilogue: bias + activation + store (C/D layout: VGPR r -> M = r + 8*halfsel)
    #pragma unroll
    for (int t = 0; t < NT; ++t) {
        const int col = t * 16 + nlane;
        const float bv = bias[col];
        #pragma unroll
        for (int r = 0; r < 8; ++r) {
            const int m = r + 8 * halfsel;
            float v = acc[t][r] + bv;
            if (act == 1) v = tanhf(v);
            C[(size_t)(rowbase + m) * ldc + col] = v;
        }
    }
}

// ---------------------------------------------------------------------------
// Segment-sum scatter (float4 per thread) + optional edge counting
// ---------------------------------------------------------------------------
__global__ void scatter_add_k(const float* __restrict__ feat, int ldf,
                              const int* __restrict__ src, const int* __restrict__ dst,
                              long long E, int F, float* __restrict__ sum,
                              float* __restrict__ cnt)
{
    const long long gid = (long long)blockIdx.x * blockDim.x + threadIdx.x;
    const int groups = F >> 2;
    if (gid >= E * groups) return;
    const long long e = gid / groups;
    const int q = (int)(gid - e * groups) * 4;
    const int s = src[e], d = dst[e];
    const float4 v = *(const float4*)(feat + (size_t)s * ldf + q);
    float* o = sum + (size_t)d * F + q;
    atomicAdd(o + 0, v.x); atomicAdd(o + 1, v.y);
    atomicAdd(o + 2, v.z); atomicAdd(o + 3, v.w);
    if (cnt && q == 0) atomicAdd(cnt + d, 1.0f);
}

__global__ void mean_div_k(float* __restrict__ sum, const float* __restrict__ cnt,
                           long long total, int F)
{
    const long long gid = (long long)blockIdx.x * blockDim.x + threadIdx.x;
    if (gid >= total) return;
    sum[gid] /= fmaxf(cnt[gid / F], 1.0f);
}

// ---------------------------------------------------------------------------
// BatchNorm (biased var, per reference) as two passes over N x 128
// ---------------------------------------------------------------------------
__global__ void bn_stats_k(const float* __restrict__ X, int M, float* __restrict__ stats)
{
    const int col    = threadIdx.x & 127;
    const int rstart = blockIdx.x * 2 + (threadIdx.x >> 7);
    const int stride = gridDim.x * 2;
    float s = 0.0f, q = 0.0f;
    for (int r = rstart; r < M; r += stride) {
        const float v = X[(size_t)r * 128 + col];
        s += v; q += v * v;
    }
    atomicAdd(&stats[col], s);
    atomicAdd(&stats[128 + col], q);
}

__global__ void bn_relu_k(const float* __restrict__ X, const float* __restrict__ stats,
                          const float* __restrict__ g, const float* __restrict__ be,
                          float* __restrict__ Y, long long total, float invM)
{
    const long long gid = (long long)blockIdx.x * blockDim.x + threadIdx.x;
    if (gid >= total) return;
    const int col = (int)(gid & 127);
    const float mu  = stats[col] * invM;
    const float var = stats[128 + col] * invM - mu * mu;
    const float y = (X[gid] - mu) * rsqrtf(var + EPSV) * g[col] + be[col];
    Y[gid] = fmaxf(y, 0.0f);
}

// ---------------------------------------------------------------------------
// Final 128 -> 1 projection + sigmoid (one row per thread)
// ---------------------------------------------------------------------------
__global__ void prob_k(const float* __restrict__ H, const float* __restrict__ w,
                       const float* __restrict__ b, float* __restrict__ out, int M)
{
    const int r = blockIdx.x * blockDim.x + threadIdx.x;
    if (r >= M) return;
    float acc = b[0];
    const float* h = H + (size_t)r * 128;
    #pragma unroll
    for (int j = 0; j < 128; j += 4) {
        const float4 v = *(const float4*)(h + j);
        acc += v.x * w[j] + v.y * w[j + 1] + v.z * w[j + 2] + v.w * w[j + 3];
    }
    out[r] = 1.0f / (1.0f + expf(-acc));
}

// ---------------------------------------------------------------------------
extern "C" void kernel_launch(void* const* d_in, const int* in_sizes, int n_in,
                              void* d_out, int out_size, void* d_ws, size_t ws_size,
                              hipStream_t stream)
{
    (void)in_sizes; (void)n_in; (void)out_size; (void)ws_size;

    const float* x    = (const float*)d_in[0];
    const int*   pe   = (const int*)d_in[1];
    const int*   ne_  = (const int*)d_in[2];
    const float* W_in = (const float*)d_in[3];  const float* b_in = (const float*)d_in[4];
    const float* W1p  = (const float*)d_in[5];  const float* b1p  = (const float*)d_in[6];
    const float* W1n  = (const float*)d_in[7];  const float* b1n  = (const float*)d_in[8];
    const float* W2p  = (const float*)d_in[9];  const float* b2p  = (const float*)d_in[10];
    const float* W2n  = (const float*)d_in[11]; const float* b2n  = (const float*)d_in[12];
    const float* Wz   = (const float*)d_in[13]; const float* bz   = (const float*)d_in[14];
    const float* Wm1  = (const float*)d_in[15]; const float* bm1  = (const float*)d_in[16];
    const float* g1   = (const float*)d_in[17]; const float* be1  = (const float*)d_in[18];
    const float* Wm2  = (const float*)d_in[19]; const float* bm2  = (const float*)d_in[20];
    const float* g2   = (const float*)d_in[21]; const float* be2  = (const float*)d_in[22];
    const float* Wm3  = (const float*)d_in[23]; const float* bm3  = (const float*)d_in[24];

    const int M = NROWS;
    const long long EP = 400000, EN = 200000;
    const int* p_src = pe;  const int* p_dst = pe  + EP;
    const int* n_src = ne_; const int* n_dst = ne_ + EN;

    float* ws = (float*)d_ws;
    const size_t SZ = (size_t)M * 128;
    float* A    = ws;                      // h -> z2 -> h2
    float* B    = ws + SZ;                 // ap -> pp|nn -> t1/t2
    float* Cb   = ws + 2 * SZ;             // an -> pn|np -> h1
    float* D    = ws + 3 * SZ;             // z1 (zp | zn)
    float* cntp = ws + 4 * SZ;
    float* cntn = cntp + M;
    float* stats = cntn + M;

    float* zout = (float*)d_out;           // N x 128
    float* prob = zout + SZ;               // N

    const unsigned GB = (unsigned)(((M / 16) + 3) / 4);   // 2344 blocks (4 waves each)
    dim3 blk(128);
    dim3 blk256(256);

    // 1. h = x @ W_in + b_in
    gemm_wmma<8><<<GB, blk, 0, stream>>>(x, 64, 64, nullptr, 0, 0, nullptr, 0, 0,
                                         W_in, b_in, A, 128, M, 0);
    // 2. ap/an = segment-mean of h (counts computed once, reused later)
    (void)hipMemsetAsync(B,  0, SZ * 4, stream);
    (void)hipMemsetAsync(Cb, 0, SZ * 4, stream);
    (void)hipMemsetAsync(cntp, 0, (size_t)M * 4, stream);
    (void)hipMemsetAsync(cntn, 0, (size_t)M * 4, stream);
    scatter_add_k<<<(unsigned)((EP * 32 + 255) / 256), blk256, 0, stream>>>(A, 128, p_src, p_dst, EP, 128, B,  cntp);
    scatter_add_k<<<(unsigned)((EN * 32 + 255) / 256), blk256, 0, stream>>>(A, 128, n_src, n_dst, EN, 128, Cb, cntn);
    {
        const long long tot = (long long)M * 128;
        mean_div_k<<<(unsigned)((tot + 255) / 256), blk256, 0, stream>>>(B,  cntp, tot, 128);
        mean_div_k<<<(unsigned)((tot + 255) / 256), blk256, 0, stream>>>(Cb, cntn, tot, 128);
    }
    // 3. z1 = tanh([ [ap|h]@W1p , [an|h]@W1n ])  -> D (zp = D, zn = D+64, ld 128)
    gemm_wmma<4><<<GB, blk, 0, stream>>>(B, 128, 128, A, 128, 128, nullptr, 0, 0,
                                         W1p, b1p, D, 128, M, 1);
    gemm_wmma<4><<<GB, blk, 0, stream>>>(Cb, 128, 128, A, 128, 128, nullptr, 0, 0,
                                         W1n, b1n, D + 64, 128, M, 1);
    // 4. pp/nn/pn/np = segment-means of zp/zn (reuse counts)
    (void)hipMemsetAsync(B,  0, SZ * 4, stream);
    (void)hipMemsetAsync(Cb, 0, SZ * 4, stream);
    scatter_add_k<<<(unsigned)((EP * 16 + 255) / 256), blk256, 0, stream>>>(D,      128, p_src, p_dst, EP, 64, B,           nullptr); // pp
    scatter_add_k<<<(unsigned)((EN * 16 + 255) / 256), blk256, 0, stream>>>(D + 64, 128, n_src, n_dst, EN, 64, B + SZ / 2,  nullptr); // nn
    scatter_add_k<<<(unsigned)((EP * 16 + 255) / 256), blk256, 0, stream>>>(D + 64, 128, p_src, p_dst, EP, 64, Cb,          nullptr); // pn
    scatter_add_k<<<(unsigned)((EN * 16 + 255) / 256), blk256, 0, stream>>>(D,      128, n_src, n_dst, EN, 64, Cb + SZ / 2, nullptr); // np
    {
        const long long tot = (long long)M * 64;
        mean_div_k<<<(unsigned)((tot + 255) / 256), blk256, 0, stream>>>(B,           cntp, tot, 64);
        mean_div_k<<<(unsigned)((tot + 255) / 256), blk256, 0, stream>>>(B + SZ / 2,  cntn, tot, 64);
        mean_div_k<<<(unsigned)((tot + 255) / 256), blk256, 0, stream>>>(Cb,          cntp, tot, 64);
        mean_div_k<<<(unsigned)((tot + 255) / 256), blk256, 0, stream>>>(Cb + SZ / 2, cntn, tot, 64);
    }
    // 5. z2 = tanh([ [pp|nn|zp]@W2p , [pn|np|zn]@W2n ])  -> A
    gemm_wmma<4><<<GB, blk, 0, stream>>>(B, 64, 64, B + SZ / 2, 64, 64, D, 64, 128,
                                         W2p, b2p, A, 128, M, 1);
    gemm_wmma<4><<<GB, blk, 0, stream>>>(Cb, 64, 64, Cb + SZ / 2, 64, 64, D + 64, 64, 128,
                                         W2n, b2n, A + 64, 128, M, 1);
    // 6. z = tanh(z2 @ Wz + bz) -> d_out (output 0)
    gemm_wmma<8><<<GB, blk, 0, stream>>>(A, 128, 128, nullptr, 0, 0, nullptr, 0, 0,
                                         Wz, bz, zout, 128, M, 1);
    // 7. h1 = relu(BN(z @ Wm1 + bm1))
    gemm_wmma<8><<<GB, blk, 0, stream>>>(zout, 128, 128, nullptr, 0, 0, nullptr, 0, 0,
                                         Wm1, bm1, B, 128, M, 0);
    (void)hipMemsetAsync(stats, 0, 256 * 4, stream);
    bn_stats_k<<<256, blk256, 0, stream>>>(B, M, stats);
    {
        const long long tot = (long long)M * 128;
        bn_relu_k<<<(unsigned)((tot + 255) / 256), blk256, 0, stream>>>(B, stats, g1, be1, Cb, tot, 1.0f / M);
    }
    // 8. h2 = relu(BN(h1 @ Wm2 + bm2))
    gemm_wmma<8><<<GB, blk, 0, stream>>>(Cb, 128, 128, nullptr, 0, 0, nullptr, 0, 0,
                                         Wm2, bm2, B, 128, M, 0);
    (void)hipMemsetAsync(stats, 0, 256 * 4, stream);
    bn_stats_k<<<256, blk256, 0, stream>>>(B, M, stats);
    {
        const long long tot = (long long)M * 128;
        bn_relu_k<<<(unsigned)((tot + 255) / 256), blk256, 0, stream>>>(B, stats, g2, be2, A, tot, 1.0f / M);
    }
    // 9. prob = sigmoid(h2 @ Wm3 + bm3) -> d_out tail (output 1)
    prob_k<<<(unsigned)((M + 255) / 256), blk256, 0, stream>>>(A, Wm3, bm3, prob, M);
}
